// TorchLikeMHA_81793357185832
// MI455X (gfx1250) — matrix operand order repaired and checked by reading, not compiled, not run on hardware
//
#include <hip/hip_runtime.h>

typedef __bf16 bf16_t;
typedef __attribute__((ext_vector_type(8)))  __bf16 v8bf;
typedef __attribute__((ext_vector_type(16))) __bf16 v16bf;
typedef __attribute__((ext_vector_type(8)))  float  v8f;

#define DIMN  1024
#define NHEAD 16
#define HDIM  64
#define BATCH 4
#define SEQ   2048
#define ROWS  (BATCH * SEQ)   // 8192

#if __has_builtin(__builtin_amdgcn_global_load_async_to_lds_b128) && \
    __has_builtin(__builtin_amdgcn_s_wait_asynccnt)
#define USE_ASYNC_LDS 1
typedef int v4i_vs __attribute__((vector_size(16)));
typedef __attribute__((address_space(1))) v4i_vs* as1_v4i;
typedef __attribute__((address_space(3))) v4i_vs* as3_v4i;
#else
#define USE_ASYNC_LDS 0
#endif

static __device__ __forceinline__ bf16_t f2bf(float f) {
  unsigned u = __builtin_bit_cast(unsigned, f);
  unsigned r = (u + 0x7FFFu + ((u >> 16) & 1u)) >> 16;
  unsigned short s = (unsigned short)r;
  return __builtin_bit_cast(bf16_t, s);
}

// Pack two floats into one dword of round-to-nearest-even bf16 pairs.
static __device__ __forceinline__ unsigned pk2(float a, float b) {
  unsigned ua = __builtin_bit_cast(unsigned, a);
  unsigned ub = __builtin_bit_cast(unsigned, b);
  ua = ua + 0x7FFFu + ((ua >> 16) & 1u);
  ub = ub + 0x7FFFu + ((ub >> 16) & 1u);
  return (ua >> 16) | (ub & 0xFFFF0000u);
}

static __device__ __forceinline__ v16bf cat16(v8bf a, v8bf b) {
  return __builtin_shufflevector(a, b, 0,1,2,3,4,5,6,7,8,9,10,11,12,13,14,15);
}

static __device__ __forceinline__ v8f wmma_bf16(v16bf a, v16bf b, v8f c) {
  return __builtin_amdgcn_wmma_f32_16x16x32_bf16(false, a, false, b, (short)0, c,
                                                 false, false);
}

static __device__ __forceinline__ float rmax16(float v) {
  v = fmaxf(v, __shfl_xor(v, 1, 32));
  v = fmaxf(v, __shfl_xor(v, 2, 32));
  v = fmaxf(v, __shfl_xor(v, 4, 32));
  v = fmaxf(v, __shfl_xor(v, 8, 32));
  return v;
}
static __device__ __forceinline__ float rsum16(float v) {
  v += __shfl_xor(v, 1, 32);
  v += __shfl_xor(v, 2, 32);
  v += __shfl_xor(v, 4, 32);
  v += __shfl_xor(v, 8, 32);
  return v;
}

// ---------------------------------------------------------------------------
// Kernel 1: fused QKV projection. Block tile 128x128, wave tile 64x32
// (8 WMMAs / 12 ds_load_b128 per K-chunk per wave).
//   grid = (ROWS/128, 24); blockIdx.y: [0,8) -> Q, [8,16) -> K, [16,24) -> V
//   Q/K layout: bf16 [B, NHEAD, SEQ, HDIM] (Q pre-scaled by 1/8).
//   V layout:   bf16 [B, NHEAD, HDIM, SEQ]  (transposed for direct PV B-frags)
// ---------------------------------------------------------------------------
__global__ __launch_bounds__(256) void qkv_kernel(
    const float* __restrict__ X,
    const float* __restrict__ Wq, const float* __restrict__ bq,
    const float* __restrict__ Wk, const float* __restrict__ bk,
    const float* __restrict__ Wv, const float* __restrict__ bv,
    bf16_t* __restrict__ qdst, bf16_t* __restrict__ kdst,
    bf16_t* __restrict__ vtdst) {
  __shared__ __align__(16) bf16_t Abuf[128][40];
  __shared__ __align__(16) bf16_t Bt[128][40];

  const int tid  = threadIdx.x;
  const int lane = tid & 31;
  const int wave = tid >> 5;
  const int wm = wave >> 2, wn = wave & 3;
  const int m0 = blockIdx.x * 128;
  const int matsel = blockIdx.y >> 3;
  const int n0 = (blockIdx.y & 7) * 128;

  const float* W    = (matsel == 0) ? Wq : ((matsel == 1) ? Wk : Wv);
  const float* bias = (matsel == 0) ? bq : ((matsel == 1) ? bk : bv);
  const float scale = (matsel == 0) ? 0.125f : 1.0f;  // 1/sqrt(64) into Q

  v8f acc[4][2];
#pragma unroll
  for (int ms = 0; ms < 4; ++ms)
#pragma unroll
    for (int ns = 0; ns < 2; ++ns)
#pragma unroll
      for (int i = 0; i < 8; ++i) acc[ms][ns][i] = 0.0f;

  const int arow  = tid >> 1;           // 0..127
  const int ahalf = (tid & 1) * 16;     // 0 / 16
  const int bc    = tid & 127;          // B column 0..127
  const int bkg   = (tid >> 7) * 16;    // B k-group 0 / 16
  const int ks8   = (lane >> 4) * 8;

  for (int k0 = 0; k0 < DIMN; k0 += 32) {
    __syncthreads();
    // Stage A: 16 f32 per thread -> 2 packed b128 LDS stores
    const float4* ap =
        (const float4*)(X + (size_t)(m0 + arow) * DIMN + k0 + ahalf);
    float4 f0 = ap[0], f1 = ap[1], f2 = ap[2], f3 = ap[3];
    uint4 ua = {pk2(f0.x, f0.y), pk2(f0.z, f0.w), pk2(f1.x, f1.y),
                pk2(f1.z, f1.w)};
    uint4 ub = {pk2(f2.x, f2.y), pk2(f2.z, f2.w), pk2(f3.x, f3.y),
                pk2(f3.z, f3.w)};
    *(uint4*)&Abuf[arow][ahalf]     = ua;
    *(uint4*)&Abuf[arow][ahalf + 8] = ub;

    // Stage B transposed: thread owns one column, 16 k-rows (lane-coalesced
    // loads per row), 2 packed b128 LDS stores.
    const float* wp = W + (size_t)(k0 + bkg) * DIMN + n0 + bc;
    float w[16];
#pragma unroll
    for (int i = 0; i < 16; ++i) w[i] = wp[i * DIMN];
    uint4 u0 = {pk2(w[0], w[1]), pk2(w[2], w[3]), pk2(w[4], w[5]),
                pk2(w[6], w[7])};
    uint4 u1 = {pk2(w[8], w[9]), pk2(w[10], w[11]), pk2(w[12], w[13]),
                pk2(w[14], w[15])};
    *(uint4*)&Bt[bc][bkg]     = u0;
    *(uint4*)&Bt[bc][bkg + 8] = u1;
    __syncthreads();

    // Batch all fragment loads, then issue WMMAs back-to-back.
    const int kb = (lane >> 4) * 16;
    v16bf Bf[2], Af[4];
#pragma unroll
    for (int ns = 0; ns < 2; ++ns) {
      const int cl = wn * 32 + ns * 16 + (lane & 15);
      Bf[ns] = cat16(*(const v8bf*)&Bt[cl][kb], *(const v8bf*)&Bt[cl][kb + 8]);
    }
#pragma unroll
    for (int ms = 0; ms < 4; ++ms) {
      const int rl = wm * 64 + ms * 16 + (lane & 15);
      Af[ms] =
          cat16(*(const v8bf*)&Abuf[rl][kb], *(const v8bf*)&Abuf[rl][kb + 8]);
    }
#pragma unroll
    for (int ms = 0; ms < 4; ++ms)
#pragma unroll
      for (int ns = 0; ns < 2; ++ns)
        acc[ms][ns] = wmma_bf16(Af[ms], Bf[ns], acc[ms][ns]);
  }

#pragma unroll
  for (int ns = 0; ns < 2; ++ns) {
    const int gcol = n0 + wn * 32 + ns * 16 + (lane & 15);
    const int h = gcol >> 6, dcol = gcol & 63;
    const float bsv = bias[gcol];

    if (matsel == 2) {
      // V^T epilogue: rows i are consecutive s -> packed b128 global stores.
#pragma unroll
      for (int ms = 0; ms < 4; ++ms) {
        const int grow0 = m0 + wm * 64 + ms * 16 + ((lane >> 4) << 3);
        const int b = grow0 >> 11, s0 = grow0 & (SEQ - 1);
        uint4 U = {pk2(acc[ms][ns][0] + bsv, acc[ms][ns][1] + bsv),
                   pk2(acc[ms][ns][2] + bsv, acc[ms][ns][3] + bsv),
                   pk2(acc[ms][ns][4] + bsv, acc[ms][ns][5] + bsv),
                   pk2(acc[ms][ns][6] + bsv, acc[ms][ns][7] + bsv)};
        *(uint4*)&vtdst[((size_t)(b * NHEAD + h) * HDIM + dcol) * SEQ + s0] = U;
      }
    } else {
      bf16_t* dst = (matsel == 0) ? qdst : kdst;
#pragma unroll
      for (int ms = 0; ms < 4; ++ms) {
#pragma unroll
        for (int i = 0; i < 8; ++i) {
          const int grow = m0 + wm * 64 + ms * 16 + i + ((lane >> 4) << 3);
          const int b = grow >> 11, s = grow & (SEQ - 1);
          const float v = (acc[ms][ns][i] + bsv) * scale;
          dst[(((size_t)(b * NHEAD + h)) * SEQ + s) * HDIM + dcol] = f2bf(v);
        }
      }
    }
  }
}

// ---------------------------------------------------------------------------
// Kernel 2: flash attention. grid = (SEQ/64, NHEAD, BATCH), 128 threads.
// Waves fully independent (no barriers): K and V^T fragments load straight
// from global; only the P C->A layout round-trip uses wave-private LDS.
// ---------------------------------------------------------------------------
__global__ __launch_bounds__(128) void attn_kernel(
    const bf16_t* __restrict__ qb, const bf16_t* __restrict__ kbf,
    const bf16_t* __restrict__ vt, const unsigned char* __restrict__ mask,
    bf16_t* __restrict__ ob) {
  __shared__ __align__(16) bf16_t Pbuf[4][16][40];

  const int tid = threadIdx.x, lane = tid & 31, wave = tid >> 5;
  const int qt = blockIdx.x, h = blockIdx.y, b = blockIdx.z;
  const size_t basebh = (size_t)(b * NHEAD + h) * SEQ;
  const int ks8 = (lane >> 4) * 8;

  // Q fragments (A layout), 16 rows x 64 d = two K=32 chunks
  const int qrow = qt * 64 + wave * 16 + (lane & 15);
  v16bf Qa[2];
#pragma unroll
  for (int c = 0; c < 2; ++c) {
    const bf16_t* qp = qb + (basebh + qrow) * HDIM + 32 * c;
    Qa[c] = cat16(*(const v8bf*)(qp + 2 * ks8), *(const v8bf*)(qp + 2 * ks8 + 8));
  }

  v8f O[4];
  float m_i[8], l_i[8];
#pragma unroll
  for (int n = 0; n < 4; ++n)
#pragma unroll
    for (int i = 0; i < 8; ++i) O[n][i] = 0.0f;
#pragma unroll
  for (int i = 0; i < 8; ++i) { m_i[i] = -3.0e38f; l_i[i] = 0.0f; }

  for (int kt = 0; kt < SEQ / 32; ++kt) {
    // Batch-load all four K fragments, then the four score WMMAs.
    v16bf Kf[2][2];
#pragma unroll
    for (int j = 0; j < 2; ++j) {
      const int keyn = kt * 32 + j * 16 + (lane & 15);
#pragma unroll
      for (int c = 0; c < 2; ++c) {
        const bf16_t* kp =
            kbf + (basebh + keyn) * HDIM + 32 * c + ((lane >> 4) << 4);
        Kf[j][c] = cat16(*(const v8bf*)kp, *(const v8bf*)(kp + 8));
      }
    }
    v8f sf[2];
#pragma unroll
    for (int j = 0; j < 2; ++j) {
      v8f s;
#pragma unroll
      for (int i = 0; i < 8; ++i) s[i] = 0.0f;
#pragma unroll
      for (int c = 0; c < 2; ++c) s = wmma_bf16(Qa[c], Kf[j][c], s);
      if (mask[b * SEQ + kt * 32 + j * 16 + (lane & 15)]) {
#pragma unroll
        for (int i = 0; i < 8; ++i) s[i] = -1.0e30f;
      }
      sf[j] = s;
    }

    // Online softmax (rows live in 16-lane halves; VGPR i <-> row i / i+8)
    float al[8];
#pragma unroll
    for (int i = 0; i < 8; ++i) {
      float mx = rmax16(fmaxf(sf[0][i], sf[1][i]));
      const float mn = fmaxf(m_i[i], mx);
      const float a  = __expf(m_i[i] - mn);
      const float e0 = __expf(sf[0][i] - mn);
      const float e1 = __expf(sf[1][i] - mn);
      const float rs = rsum16(e0 + e1);
      m_i[i] = mn;
      l_i[i] = l_i[i] * a + rs;
      al[i]  = a;
      const int rl = i + ((lane >> 4) << 3);
      Pbuf[wave][rl][lane & 15]        = f2bf(e0);
      Pbuf[wave][rl][16 + (lane & 15)] = f2bf(e1);
    }
#pragma unroll
    for (int n = 0; n < 4; ++n)
#pragma unroll
      for (int i = 0; i < 8; ++i) O[n][i] *= al[i];

    // P (C layout) -> A layout via wave-private LDS, then O += P @ V
    const int prl = lane & 15;
    v16bf Pa = cat16(*(const v8bf*)&Pbuf[wave][prl][2 * ks8],
                     *(const v8bf*)&Pbuf[wave][prl][2 * ks8 + 8]);
    const int kb2 = (lane >> 4) * 16;
    v16bf Vf[4];
#pragma unroll
    for (int n = 0; n < 4; ++n) {
      const int dc = n * 16 + (lane & 15);
      const bf16_t* vp =
          vt + ((size_t)(b * NHEAD + h) * HDIM + dc) * SEQ + kt * 32 + kb2;
      Vf[n] = cat16(*(const v8bf*)vp, *(const v8bf*)(vp + 8));
    }
#pragma unroll
    for (int n = 0; n < 4; ++n) O[n] = wmma_bf16(Pa, Vf[n], O[n]);
  }

  // Normalize and store attention output as bf16 [B, S, DIMN] (heads merged)
#pragma unroll
  for (int n = 0; n < 4; ++n) {
    const int dc = n * 16 + (lane & 15);
#pragma unroll
    for (int i = 0; i < 8; ++i) {
      const int s = qt * 64 + wave * 16 + i + ((lane >> 4) << 3);
      const float v = O[n][i] / l_i[i];
      ob[((size_t)b * SEQ + s) * DIMN + h * HDIM + dc] = f2bf(v);
    }
  }
}

// ---------------------------------------------------------------------------
// Kernel 3: output projection. grid = (ROWS/128, DIMN/128), 256 threads.
// Block tile 128x128, wave tile 64x32. A (bf16) staged into LDS with async
// global->LDS DMA when available.
// ---------------------------------------------------------------------------
__global__ __launch_bounds__(256) void oproj_kernel(
    const bf16_t* __restrict__ A, const float* __restrict__ W,
    const float* __restrict__ bias, float* __restrict__ out) {
  __shared__ __align__(16) bf16_t Abuf[128][40];
  __shared__ __align__(16) bf16_t Bt[128][40];

  const int tid  = threadIdx.x;
  const int lane = tid & 31;
  const int wave = tid >> 5;
  const int wm = wave >> 2, wn = wave & 3;
  const int m0 = blockIdx.x * 128;
  const int n0 = blockIdx.y * 128;

  v8f acc[4][2];
#pragma unroll
  for (int ms = 0; ms < 4; ++ms)
#pragma unroll
    for (int ns = 0; ns < 2; ++ns)
#pragma unroll
      for (int i = 0; i < 8; ++i) acc[ms][ns][i] = 0.0f;

  const int arow  = tid >> 1;
  const int ahalf = (tid & 1) * 16;
  const int bc    = tid & 127;
  const int bkg   = (tid >> 7) * 16;
  const int ks8   = (lane >> 4) * 8;

  for (int k0 = 0; k0 < DIMN; k0 += 32) {
    __syncthreads();
    // Stage A (bf16, pure byte copy): async global->LDS DMA when available.
    const bf16_t* gp = A + (size_t)(m0 + arow) * DIMN + k0 + ahalf;
    bf16_t* lp = &Abuf[arow][ahalf];
#if USE_ASYNC_LDS
    __builtin_amdgcn_global_load_async_to_lds_b128(
        (as1_v4i)(gp), (as3_v4i)(lp), 0, 0);
    __builtin_amdgcn_global_load_async_to_lds_b128(
        (as1_v4i)(gp + 8), (as3_v4i)(lp + 8), 0, 0);
#else
    const v8bf* ap = (const v8bf*)gp;
    *(v8bf*)lp       = ap[0];
    *(v8bf*)(lp + 8) = ap[1];
#endif

    // Stage B transposed (f32 -> packed bf16), 2 b128 LDS stores per thread.
    const float* wp = W + (size_t)(k0 + bkg) * DIMN + n0 + bc;
    float w[16];
#pragma unroll
    for (int i = 0; i < 16; ++i) w[i] = wp[i * DIMN];
    uint4 u0 = {pk2(w[0], w[1]), pk2(w[2], w[3]), pk2(w[4], w[5]),
                pk2(w[6], w[7])};
    uint4 u1 = {pk2(w[8], w[9]), pk2(w[10], w[11]), pk2(w[12], w[13]),
                pk2(w[14], w[15])};
    *(uint4*)&Bt[bc][bkg]     = u0;
    *(uint4*)&Bt[bc][bkg + 8] = u1;

#if USE_ASYNC_LDS
    __builtin_amdgcn_s_wait_asynccnt(0);
#endif
    __syncthreads();

    const int kb = (lane >> 4) * 16;
    v16bf Bf[2], Af[4];
#pragma unroll
    for (int ns = 0; ns < 2; ++ns) {
      const int cl = wn * 32 + ns * 16 + (lane & 15);
      Bf[ns] = cat16(*(const v8bf*)&Bt[cl][kb], *(const v8bf*)&Bt[cl][kb + 8]);
    }
#pragma unroll
    for (int ms = 0; ms < 4; ++ms) {
      const int rl = wm * 64 + ms * 16 + (lane & 15);
      Af[ms] =
          cat16(*(const v8bf*)&Abuf[rl][kb], *(const v8bf*)&Abuf[rl][kb + 8]);
    }
#pragma unroll
    for (int ms = 0; ms < 4; ++ms)
#pragma unroll
      for (int ns = 0; ns < 2; ++ns)
        acc[ms][ns] = wmma_bf16(Af[ms], Bf[ns], acc[ms][ns]);
  }

#pragma unroll
  for (int ns = 0; ns < 2; ++ns) {
    const int gcol = n0 + wn * 32 + ns * 16 + (lane & 15);
    const float bsv = bias[gcol];
#pragma unroll
    for (int ms = 0; ms < 4; ++ms) {
#pragma unroll
      for (int i = 0; i < 8; ++i) {
        const int grow = m0 + wm * 64 + ms * 16 + i + ((lane >> 4) << 3);
        out[(size_t)grow * DIMN + gcol] = acc[ms][ns][i] + bsv;
      }
    }
  }
}

extern "C" void kernel_launch(void* const* d_in, const int* in_sizes, int n_in,
                              void* d_out, int out_size, void* d_ws,
                              size_t ws_size, hipStream_t stream) {
  (void)in_sizes; (void)n_in; (void)out_size; (void)ws_size;
  const float* X  = (const float*)d_in[0];
  const unsigned char* mask = (const unsigned char*)d_in[1];  // jax bool -> u8
  const float* Wq = (const float*)d_in[2];
  const float* bq = (const float*)d_in[3];
  const float* Wk = (const float*)d_in[4];
  const float* bk = (const float*)d_in[5];
  const float* Wv = (const float*)d_in[6];
  const float* bv = (const float*)d_in[7];
  const float* Wo = (const float*)d_in[8];
  const float* bo = (const float*)d_in[9];
  float* out = (float*)d_out;

  const size_t nelem = (size_t)ROWS * DIMN;
  bf16_t* qb = (bf16_t*)d_ws;
  bf16_t* kb = qb + nelem;
  bf16_t* vt = kb + nelem;   // V^T: [B, NHEAD, HDIM, SEQ]
  bf16_t* ab = vt + nelem;   // total workspace: 4 * 16 MB = 64 MB

  qkv_kernel<<<dim3(ROWS / 128, 24), 256, 0, stream>>>(X, Wq, bq, Wk, bk, Wv,
                                                       bv, qb, kb, vt);
  attn_kernel<<<dim3(SEQ / 64, NHEAD, BATCH), 128, 0, stream>>>(qb, kb, vt,
                                                                mask, ab);
  oproj_kernel<<<dim3(ROWS / 128, DIMN / 128), 256, 0, stream>>>(ab, Wo, bo,
                                                                 out);
}
